// BiLSTM_1_24309514895773
// MI455X (gfx1250) — compile-verified, matrix operand order checked by reading, not run on hardware
//
#include <hip/hip_runtime.h>
#include <cmath>

typedef __attribute__((ext_vector_type(16))) __bf16 v16bf;
typedef __attribute__((ext_vector_type(8)))  __bf16 v8bf;
typedef __attribute__((ext_vector_type(8)))  float  v8f;

__device__ __forceinline__ v16bf mk16(v8bf lo, v8bf hi) {
  v16bf r;
#pragma unroll
  for (int e = 0; e < 8; ++e) { r[e] = lo[e]; r[8 + e] = hi[e]; }
  return r;
}

__device__ __forceinline__ v8f wmma_bf16(v16bf a, v16bf b, v8f c) {
  return __builtin_amdgcn_wmma_f32_16x16x32_bf16(false, a, false, b, (short)0, c, false, false);
}

__device__ __forceinline__ float sigmoidf_(float x) {
  return 1.0f / (1.0f + __expf(-x));
}

// ---------------------------------------------------------------------------
// Pack a [K, N] f32 row-major matrix into bf16 WMMA B-fragment order:
//   P[((kt*nnt + nt)*32 + lane)*16 + e] = W[(kt*32 + Klocal(e,lane))*N + nt*16 + lane%16]
//   Klocal = (e<8 ? e : e+8) + 8*(lane>=16)
// ---------------------------------------------------------------------------
__global__ void pack_w(const float* __restrict__ W, __bf16* __restrict__ P,
                       int Ktot, int N) {
  int idx = blockIdx.x * 256 + threadIdx.x;
  int total = Ktot * N;
  if (idx >= total) return;
  int e    = idx & 15;
  int lane = (idx >> 4) & 31;
  int frag = idx >> 9;
  int nnt  = N >> 4;
  int kt   = frag / nnt;
  int nt   = frag % nnt;
  int g    = lane >> 4;
  int kl   = (e < 8 ? e : e + 8) + 8 * g;
  int Kr   = kt * 32 + kl;
  int col  = nt * 16 + (lane & 15);
  P[idx] = (__bf16)W[(size_t)Kr * N + col];
}

__global__ void f32_to_bf16(const float* __restrict__ in, __bf16* __restrict__ out, int n) {
  int i = blockIdx.x * 256 + threadIdx.x;
  if (i < n) out[i] = (__bf16)in[i];
}

__global__ void highway_blend(const float* __restrict__ gpre, const float* __restrict__ outv,
                              const float* __restrict__ cur, float* __restrict__ dst, int n) {
  int i = blockIdx.x * 256 + threadIdx.x;
  if (i < n) {
    float gg = sigmoidf_(gpre[i]);
    dst[i] = gg * outv[i] + (1.0f - gg) * cur[i];
  }
}

// ---------------------------------------------------------------------------
// C[M,N] = A[M,K](bf16 row-major) @ Bpacked + bias.  WG = 256 thr (8 waves,
// 2x4 wave grid), block tile 64(M) x 256(N), K step 32.  A staged through LDS
// with register prefetch of the next K-chunk; B fragments prefetched to L2.
// ---------------------------------------------------------------------------
__global__ __launch_bounds__(256) void gemm_bf16_wmma(
    const __bf16* __restrict__ A, const __bf16* __restrict__ Bp,
    const float* __restrict__ bias, float* __restrict__ C,
    int M, int N, int K) {
  __shared__ __bf16 As[64 * 32];
  const int tid  = threadIdx.x;
  const int lane = tid & 31;
  const int w    = tid >> 5;
  const int wm   = w >> 2;
  const int wn   = w & 3;
  const int g    = lane >> 4;
  const int lr   = lane & 15;
  const int m0   = blockIdx.x * 64;
  const int ntg0 = blockIdx.y * 16 + wn * 4;
  const int nnt  = N >> 4;
  const int nkt  = K >> 5;

  v8f acc[2][4];
#pragma unroll
  for (int n = 0; n < 4; ++n) {
    const float bv = bias ? bias[(ntg0 + n) * 16 + lr] : 0.0f;
#pragma unroll
    for (int m = 0; m < 2; ++m)
#pragma unroll
      for (int r = 0; r < 8; ++r) acc[m][n][r] = bv;
  }

  const int srow   = tid >> 2;
  const int schunk = (tid & 3) * 8;
  const __bf16* arow = A + (size_t)(m0 + srow) * K + schunk;

  v8bf areg = *(const v8bf*)arow;  // prefetch kt=0 chunk into registers

#pragma unroll 1
  for (int kt = 0; kt < nkt; ++kt) {
    __syncthreads();  // previous iteration's LDS readers done
    *(v8bf*)&As[srow * 32 + schunk] = areg;
    __syncthreads();

    if (kt + 1 < nkt) {
      areg = *(const v8bf*)(arow + (kt + 1) * 32);  // overlap with WMMA below
#pragma unroll
      for (int n = 0; n < 4; ++n)
        __builtin_prefetch(
            Bp + ((((size_t)(kt + 1) * nnt + (ntg0 + n)) * 32 + lane) << 4), 0, 3);
    }

    const __bf16* ap0 = &As[(wm * 32 + lr) * 32 + 8 * g];
    v16bf a0 = mk16(*(const v8bf*)ap0, *(const v8bf*)(ap0 + 16));
    const __bf16* ap1 = &As[(wm * 32 + 16 + lr) * 32 + 8 * g];
    v16bf a1 = mk16(*(const v8bf*)ap1, *(const v8bf*)(ap1 + 16));

#pragma unroll
    for (int n = 0; n < 4; ++n) {
      const v8bf* q = (const v8bf*)(Bp + ((((size_t)kt * nnt + (ntg0 + n)) * 32 + lane) << 4));
      v16bf bfrag = mk16(q[0], q[1]);
      acc[0][n] = wmma_bf16(a0, bfrag, acc[0][n]);
      acc[1][n] = wmma_bf16(a1, bfrag, acc[1][n]);
    }
  }

#pragma unroll
  for (int m = 0; m < 2; ++m)
#pragma unroll
    for (int n = 0; n < 4; ++n)
#pragma unroll
      for (int r = 0; r < 8; ++r)
        C[(size_t)(m0 + wm * 32 + m * 16 + r + 8 * g) * N + (ntg0 + n) * 16 + lr] =
            acc[m][n][r];
}

// ---------------------------------------------------------------------------
// Persistent recurrent scan: grid = 2 (dir), 256 threads (8 waves).
// z_t = Zx[t] + h@Wh ; i=sig, f=1-i, c' = (1-i)c + i tanh(j), h' = tanh(c') sig(o)
// h kept bf16 in LDS (16KB), c in VGPRs, Wh bf16 packed fragments from L2.
// Next step's Zx prefetched during the WMMA phase.
// ---------------------------------------------------------------------------
__global__ __launch_bounds__(256) void lstm_scan(
    const float* __restrict__ Zf, const float* __restrict__ Zb,
    const __bf16* __restrict__ Whf, const __bf16* __restrict__ Whb,
    const float* __restrict__ h0, const float* __restrict__ c0,
    float* __restrict__ y, float* __restrict__ hn, float* __restrict__ cn) {
  const int dir = blockIdx.x;
  const float*  Zx = dir ? Zb : Zf;
  const __bf16* Wh = dir ? Whb : Whf;

  __shared__ __bf16 hshm[32 * 256];

  const int tid  = threadIdx.x;
  const int lane = tid & 31;
  const int w    = tid >> 5;
  const int g    = lane >> 4;
  const int lr   = lane & 15;

  for (int i = tid; i < 32 * 256; i += 256) hshm[i] = (__bf16)h0[i & 255];

  v8f creg[2][2];
#pragma unroll
  for (int hti = 0; hti < 2; ++hti) {
    const int ht = 2 * w + hti;
    const float cv = c0[ht * 16 + lr];
#pragma unroll
    for (int m = 0; m < 2; ++m)
#pragma unroll
      for (int r = 0; r < 8; ++r) creg[hti][m][r] = cv;
  }
  __syncthreads();

#pragma unroll 1
  for (int s = 0; s < 1024; ++s) {
    const int t = dir ? (1023 - s) : s;
    const float* Zt = Zx + (size_t)t * (32 * 768);

    v8f z[2][3][2];
#pragma unroll
    for (int hti = 0; hti < 2; ++hti) {
      const int ht = 2 * w + hti;
#pragma unroll
      for (int gate = 0; gate < 3; ++gate) {
        const int ntg = gate * 16 + ht;
#pragma unroll
        for (int m = 0; m < 2; ++m)
#pragma unroll
          for (int r = 0; r < 8; ++r)
            z[hti][gate][m][r] = Zt[(size_t)(m * 16 + r + 8 * g) * 768 + ntg * 16 + lr];
      }
    }

    // Prefetch next timestep's pre-activations (768 x 128B lines) into cache
    // while this step's WMMAs run.
    if (s + 1 < 1024) {
      const float* Zn = Zx + (size_t)(dir ? (1022 - s) : (s + 1)) * (32 * 768);
#pragma unroll
      for (int pi = 0; pi < 3; ++pi)
        __builtin_prefetch(Zn + ((size_t)tid * 3 + pi) * 32, 0, 3);
    }

#pragma unroll
    for (int kt = 0; kt < 8; ++kt) {
      const __bf16* ap0 = &hshm[lr * 256 + kt * 32 + 8 * g];
      v16bf a0 = mk16(*(const v8bf*)ap0, *(const v8bf*)(ap0 + 16));
      const __bf16* ap1 = &hshm[(16 + lr) * 256 + kt * 32 + 8 * g];
      v16bf a1 = mk16(*(const v8bf*)ap1, *(const v8bf*)(ap1 + 16));
#pragma unroll
      for (int hti = 0; hti < 2; ++hti) {
#pragma unroll
        for (int gate = 0; gate < 3; ++gate) {
          const int ntg = gate * 16 + 2 * w + hti;
          const v8bf* q = (const v8bf*)(Wh + ((((size_t)kt * 48 + ntg) * 32 + lane) << 4));
          v16bf bfrag = mk16(q[0], q[1]);
          z[hti][gate][0] = wmma_bf16(a0, bfrag, z[hti][gate][0]);
          z[hti][gate][1] = wmma_bf16(a1, bfrag, z[hti][gate][1]);
        }
      }
    }

    __syncthreads();  // all A-fragment reads of hshm done

#pragma unroll
    for (int hti = 0; hti < 2; ++hti) {
      const int ht  = 2 * w + hti;
      const int col = ht * 16 + lr;
#pragma unroll
      for (int m = 0; m < 2; ++m) {
#pragma unroll
        for (int r = 0; r < 8; ++r) {
          const float iv   = sigmoidf_(z[hti][0][m][r]);
          const float jv   = tanhf(z[hti][1][m][r]);
          const float ov   = sigmoidf_(z[hti][2][m][r]);
          const float cnew = (1.0f - iv) * creg[hti][m][r] + iv * jv;
          creg[hti][m][r]  = cnew;
          const float hnew = tanhf(cnew) * ov;
          const int row    = m * 16 + r + 8 * g;
          hshm[row * 256 + col] = (__bf16)hnew;
          y[((size_t)t * 32 + row) * 512 + dir * 256 + col] = hnew;
        }
      }
    }
    __syncthreads();  // new h visible before next step's reads
  }

  float* hnp = hn + (size_t)dir * 32 * 256;
  for (int i = tid; i < 32 * 256; i += 256) hnp[i] = (float)hshm[i];
  float* cnp = cn + (size_t)dir * 32 * 256;
#pragma unroll
  for (int hti = 0; hti < 2; ++hti) {
    const int ht = 2 * w + hti;
#pragma unroll
    for (int m = 0; m < 2; ++m)
#pragma unroll
      for (int r = 0; r < 8; ++r)
        cnp[(size_t)(m * 16 + r + 8 * g) * 256 + ht * 16 + lr] = creg[hti][m][r];
  }
}

// ---------------------------------------------------------------------------
extern "C" void kernel_launch(void* const* d_in, const int* in_sizes, int n_in,
                              void* d_out, int out_size, void* d_ws, size_t ws_size,
                              hipStream_t stream) {
  (void)in_sizes; (void)n_in; (void)out_size; (void)ws_size;
  const int TB = 1024 * 32;

  const float* x  = (const float*)d_in[0];
  const float* h0 = (const float*)d_in[1];
  const float* c0 = (const float*)d_in[2];
  const float* W[6];
  const float* bvec[6];
  for (int i = 0; i < 6; ++i) {
    W[i]    = (const float*)d_in[3 + 2 * i];
    bvec[i] = (const float*)d_in[4 + 2 * i];
  }
  const float* Whw = (const float*)d_in[15];
  const float* bhw = (const float*)d_in[16];

  char* wsb = (char*)d_ws;
  size_t off = 0;
  auto take = [&](size_t bytes) -> char* {
    char* p = wsb + off;
    off += (bytes + 255) & ~(size_t)255;
    return p;
  };
  __bf16* P[6];
  P[0] = (__bf16*)take((size_t)512 * 768 * 2);
  P[1] = (__bf16*)take((size_t)512 * 768 * 2);
  P[2] = (__bf16*)take((size_t)768 * 768 * 2);
  P[3] = (__bf16*)take((size_t)768 * 768 * 2);
  P[4] = (__bf16*)take((size_t)768 * 768 * 2);
  P[5] = (__bf16*)take((size_t)768 * 768 * 2);
  __bf16* Phw = (__bf16*)take((size_t)512 * 512 * 2);
  __bf16* xb  = (__bf16*)take((size_t)TB * 512 * 2);
  float* zf   = (float*)take((size_t)TB * 768 * 4);
  float* zb   = (float*)take((size_t)TB * 768 * 4);
  float* bufA = (float*)take((size_t)TB * 512 * 4);
  float* bufB = (float*)take((size_t)TB * 512 * 4);
  float* gpre = zf;  // reuse (zf consumed by scan before highway gemm)

  float* outp = (float*)d_out;
  float* hn   = outp + (size_t)TB * 512;
  float* cn   = hn + (size_t)6 * 32 * 256;

  auto pack = [&](const float* Wm, __bf16* Pm, int Kt, int Nn) {
    int total = Kt * Nn;
    pack_w<<<dim3((total + 255) / 256), dim3(256), 0, stream>>>(Wm, Pm, Kt, Nn);
  };
  pack(W[0], P[0], 512, 768);
  pack(W[1], P[1], 512, 768);
  pack(W[2], P[2], 768, 768);
  pack(W[3], P[3], 768, 768);
  pack(W[4], P[4], 768, 768);
  pack(W[5], P[5], 768, 768);
  pack(Whw, Phw, 512, 512);

  auto conv = [&](const float* src, __bf16* dst, int n) {
    f32_to_bf16<<<dim3((n + 255) / 256), dim3(256), 0, stream>>>(src, dst, n);
  };
  auto gemm = [&](const __bf16* Am, const __bf16* Bm, const float* bb, float* Cm,
                  int N, int K) {
    gemm_bf16_wmma<<<dim3(TB / 64, N / 256), dim3(256), 0, stream>>>(Am, Bm, bb, Cm, TB, N, K);
  };

  // -------- layer 0 (K = 256, Wh fragment offset = 8 kt blocks) --------
  conv(x, xb, TB * 256);
  gemm(xb, P[0], bvec[0], zf, 768, 256);
  gemm(xb, P[1], bvec[1], zb, 768, 256);
  lstm_scan<<<dim3(2), dim3(256), 0, stream>>>(zf, zb, P[0] + 8 * 48 * 512,
                                               P[1] + 8 * 48 * 512, h0, c0, bufA,
                                               hn + 0, cn + 0);
  // -------- layer 1 (K = 512, Wh offset = 16 kt blocks) --------
  conv(bufA, xb, TB * 512);
  gemm(xb, P[2], bvec[2], zf, 768, 512);
  gemm(xb, P[3], bvec[3], zb, 768, 512);
  lstm_scan<<<dim3(2), dim3(256), 0, stream>>>(zf, zb, P[2] + 16 * 48 * 512,
                                               P[3] + 16 * 48 * 512, h0, c0, bufB,
                                               hn + 2 * 8192, cn + 2 * 8192);
  conv(bufB, xb, TB * 512);
  gemm(xb, Phw, bhw, gpre, 512, 512);
  highway_blend<<<dim3((TB * 512 + 255) / 256), dim3(256), 0, stream>>>(gpre, bufB, bufA,
                                                                        bufA, TB * 512);
  // -------- layer 2 --------
  conv(bufA, xb, TB * 512);
  gemm(xb, P[4], bvec[4], zf, 768, 512);
  gemm(xb, P[5], bvec[5], zb, 768, 512);
  lstm_scan<<<dim3(2), dim3(256), 0, stream>>>(zf, zb, P[4] + 16 * 48 * 512,
                                               P[5] + 16 * 48 * 512, h0, c0, bufB,
                                               hn + 4 * 8192, cn + 4 * 8192);
  conv(bufB, xb, TB * 512);
  gemm(xb, Phw, bhw, gpre, 512, 512);
  highway_blend<<<dim3((TB * 512 + 255) / 256), dim3(256), 0, stream>>>(gpre, bufB, bufA,
                                                                        outp, TB * 512);
}
